// GATE3a_18159121728222
// MI455X (gfx1250) — compile-verified, matrix operand order checked by reading, not run on hardware
//
#include <hip/hip_runtime.h>

typedef __attribute__((ext_vector_type(16))) __bf16 v16bf;
typedef __attribute__((ext_vector_type(8)))  float  v8f;
typedef unsigned __attribute__((ext_vector_type(4))) uint32x4;
typedef int      __attribute__((ext_vector_type(8))) int32x8;
typedef int      __attribute__((ext_vector_type(4))) int32x4;

#define TILE_M 64
#define TILE_N 64
#define TILE_K 32
#define PITCH  40   // u16; row stride 80B = 20 DWORDs (16B aligned, bank-clean)

#if __has_builtin(__builtin_amdgcn_tensor_load_to_lds)
#define USE_TDM 1
#else
#define USE_TDM 0
#endif

__device__ __forceinline__ unsigned f2bf_bits(float f) {
  union { float f; unsigned u; } a; a.f = f;
  return (a.u + 0x7FFFu + ((a.u >> 16) & 1u)) >> 16;   // rne bf16 bits
}
__device__ __forceinline__ unsigned pack2(float lo, float hi) {
  return f2bf_bits(lo) | (f2bf_bits(hi) << 16);
}

__device__ __forceinline__ v16bf load_frag(const unsigned short* rowp, int kd) {
  // ISA 16-bit A/B fragment layout: lane half selects K base 0 vs 8 (kd = dword
  // offset), u[0..3] = K base..base+7, u[4..7] = K base+16..base+23
  union { v16bf v; unsigned u[8]; } f;
  const unsigned* p = (const unsigned*)rowp;
#pragma unroll
  for (int j = 0; j < 4; ++j) { f.u[j] = p[kd + j]; f.u[4 + j] = p[kd + 8 + j]; }
  return f.v;
}

#if USE_TDM
// TDM: DMA a 64(N) x 32(K) bf16 weight tile into LDS at `lds_byte`, with
// 4-DWORD padding after each 16-DWORD row -> lands in PITCH=40 u16 layout.
// OOB (K/N tails) reads return zero per ISA; LDS addressing still advances.
__device__ __forceinline__ void tdm_load_b_tile(
    const unsigned short* Wb, unsigned lds_byte, int n0, int k0, int N, int K) {
  unsigned long long ga =
      (unsigned long long)(const void*)Wb + ((size_t)n0 * K + k0) * 2ull;
  unsigned td0 = (unsigned)(K - k0);     // remaining cols (elements)
  unsigned td1 = (unsigned)(N - n0);     // remaining rows
  unsigned str = (unsigned)K;            // row stride (elements)
  uint32x4 g0;
  g0[0] = 1u;                                            // count=1, user D#
  g0[1] = lds_byte;                                      // lds_addr
  g0[2] = (unsigned)(ga & 0xFFFFFFFFu);                  // global_addr[31:0]
  g0[3] = (unsigned)((ga >> 32) & 0x1FFFFFFu) | (2u << 30); // addr[56:32]|type=2
  int32x8 g1;
  g1[0] = (int)((1u << 16)        // data_size = 2B
              | (1u << 20)        // pad_enable
              | (3u << 22)        // pad_interval: 16 DWORDs
              | (3u << 25));      // pad_amount: 4 DWORDs
  g1[1] = (int)((td0 & 0xFFFFu) << 16);                  // tensor_dim0[15:0]
  g1[2] = (int)((td0 >> 16) | ((td1 & 0xFFFFu) << 16));  // td0[31:16],td1[15:0]
  g1[3] = (int)((td1 >> 16) | (32u << 16));              // td1[31:16],tile_dim0=32
  g1[4] = (int)64u;                                      // tile_dim1=64,tile_dim2=0
  g1[5] = (int)str;                                      // dim0_stride[31:0]
  g1[6] = 0;                                             // stride hi, dim1_stride
  g1[7] = 0;
  int32x4 z4 = {0, 0, 0, 0};
  int32x8 z8 = {0, 0, 0, 0, 0, 0, 0, 0};
  __builtin_amdgcn_tensor_load_to_lds(g0, g1, z4, z4, z8, 0);
}
#endif

// Y[M,N] = act( concat(seg0,seg1,seg2)[M,K] @ Wb[N,K]^T + bias ), Wb in bf16.
// Requires: W0/W1/W2 and K multiples of 8 (true here: all multiples of 16).
// 256 threads = 8 waves; block tile 64x64; wave tile 16x32 (2 acc); K step 32.
__global__ __launch_bounds__(256)
void wmma_gemm_gather(
    const float* __restrict__ S0, const int* __restrict__ I0, int W0,
    const float* __restrict__ S1, const int* __restrict__ I1, int W1,
    const float* __restrict__ S2, const int* __restrict__ I2, int W2,
    const unsigned short* __restrict__ Wb, const float* __restrict__ bias,
    float* __restrict__ Y, int M, int N, int K, int relu)
{
  __shared__ __align__(16) unsigned short As[2][TILE_M][PITCH];
  __shared__ __align__(16) unsigned short Bs[2][TILE_N][PITCH];

  const int tid  = threadIdx.x;
  const int lane = tid & 31;
  const int wid  = tid >> 5;          // 0..7
  const int mt   = wid & 3;           // 0..3  (M sub-tile, 16 rows)
  const int nt   = wid >> 2;          // 0..1  (N sub-tile, 32 cols)
  const int m0   = blockIdx.x * TILE_M;
  const int n0   = blockIdx.y * TILE_N;

  // staging task: one 8-float chunk per thread per K-step
  const int sr = tid >> 2;            // tile row 0..63
  const int ck = (tid & 3) * 8;       // tile col 0,8,16,24
  const int gm = m0 + sr;

  // Hoist gather indices + segment base pointers out of the K loop
  const float* base0 = S0;
  const float* base1 = S1;
  const float* base2 = S2;
  if (gm < M) {
    base0 = S0 + (size_t)(I0 ? I0[gm] : gm) * W0;
    if (S1) base1 = S1 + (size_t)(I1 ? I1[gm] : gm) * W1;
    if (S2) base2 = S2 + (size_t)(I2 ? I2[gm] : gm) * W2;
  }

  auto stageA = [&](int buf, int k0) {
    const int gk = k0 + ck;
    float4 a = {0.f, 0.f, 0.f, 0.f}, b = {0.f, 0.f, 0.f, 0.f};
    if (gm < M && gk < K) {
      const float* p; int kk = gk;
      if (kk < W0)              p = base0 + kk;
      else if ((kk -= W0) < W1) p = base1 + kk;
      else                      p = base2 + (kk - W1);
      a = *(const float4*)p;
      b = *(const float4*)(p + 4);
      if (gk + TILE_K < K) __builtin_prefetch(p + TILE_K, 0, 0);
    }
    *(uint4*)&As[buf][sr][ck] =
        make_uint4(pack2(a.x, a.y), pack2(a.z, a.w),
                   pack2(b.x, b.y), pack2(b.z, b.w));
  };

#if !USE_TDM
  auto stageB = [&](int buf, int k0) {
    const int gk = k0 + ck;
    const int gn = n0 + sr;
    uint4 v = make_uint4(0u, 0u, 0u, 0u);
    if (gn < N && gk < K)
      v = *(const uint4*)(Wb + (size_t)gn * K + gk);   // 8 bf16 = 16B
    *(uint4*)&Bs[buf][sr][ck] = v;
  };
#endif

  v8f acc0 = {}, acc1 = {};
  const int rrow = lane & 15;
  const int kd   = (lane >> 4) ? 4 : 0;

  int buf = 0;
  stageA(0, 0);
#if USE_TDM
  if (wid == 0) {
    tdm_load_b_tile(Wb, (unsigned)(unsigned long long)&Bs[0][0][0],
                    n0, 0, N, K);
    __builtin_amdgcn_s_wait_tensorcnt(0);
  }
#else
  stageB(0, 0);
#endif
  __syncthreads();

  for (int k0 = 0; k0 < K; k0 += TILE_K) {
    const bool more = (k0 + TILE_K) < K;
    if (more) {
      stageA(buf ^ 1, k0 + TILE_K);            // overlap next-tile loads
#if USE_TDM
      if (wid == 0)
        tdm_load_b_tile(Wb, (unsigned)(unsigned long long)&Bs[buf ^ 1][0][0],
                        n0, k0 + TILE_K, N, K);
#else
      stageB(buf ^ 1, k0 + TILE_K);
#endif
    }
    v16bf af = load_frag(&As[buf][mt * 16 + rrow][0], kd);
    v16bf b0 = load_frag(&Bs[buf][nt * 32 + rrow][0], kd);
    v16bf b1 = load_frag(&Bs[buf][nt * 32 + 16 + rrow][0], kd);
    acc0 = __builtin_amdgcn_wmma_f32_16x16x32_bf16(
        false, af, false, b0, (short)0, acc0, false, false);
    acc1 = __builtin_amdgcn_wmma_f32_16x16x32_bf16(
        false, af, false, b1, (short)0, acc1, false, false);
#if USE_TDM
    if (wid == 0 && more) __builtin_amdgcn_s_wait_tensorcnt(0);
#endif
    __syncthreads();
    buf ^= 1;
  }

  // Writeback: lane 0-15 -> N=lane (+16 for acc1), lanes 16-31 -> rows +8
  const int nA = n0 + nt * 32 + (lane & 15);
  const int nB = nA + 16;
  const float bvA = (bias && nA < N) ? bias[nA] : 0.f;
  const float bvB = (bias && nB < N) ? bias[nB] : 0.f;
  const int mbase = m0 + mt * 16 + ((lane >> 4) ? 8 : 0);
#pragma unroll
  for (int r = 0; r < 8; ++r) {
    int m = mbase + r;
    if (m < M) {
      if (nA < N) {
        float v = acc0[r] + bvA;
        if (relu) v = fmaxf(v, 0.f);
        Y[(size_t)m * N + nA] = v;
      }
      if (nB < N) {
        float v = acc1[r] + bvB;
        if (relu) v = fmaxf(v, 0.f);
        Y[(size_t)m * N + nB] = v;
      }
    }
  }
}

__global__ __launch_bounds__(256)
void cvt_f32_bf16(const float* __restrict__ src, unsigned short* __restrict__ dst,
                  long long n) {
  long long i = (long long)blockIdx.x * blockDim.x + threadIdx.x;
  long long s = (long long)gridDim.x * blockDim.x;
  for (; i < n; i += s) dst[i] = (unsigned short)f2bf_bits(src[i]);
}

__global__ __launch_bounds__(256)
void fill_f32(float* __restrict__ p, float v, long long n) {
  long long i = (long long)blockIdx.x * blockDim.x + threadIdx.x;
  long long s = (long long)gridDim.x * blockDim.x;
  for (; i < n; i += s) p[i] = v;
}

__global__ __launch_bounds__(256)
void scatter_add(const float* __restrict__ V, const int* __restrict__ idx,
                 float* __restrict__ sums, float* __restrict__ cnt,
                 long long M, int F) {
  long long total = M * F;
  long long s = (long long)gridDim.x * blockDim.x;
  for (long long t = (long long)blockIdx.x * blockDim.x + threadIdx.x;
       t < total; t += s) {
    int m = (int)(t / F);
    int f = (int)(t - (long long)m * F);
    int d = idx[m];
    atomicAdd(&sums[(size_t)d * F + f], V[t]);
    if (f == 0) atomicAdd(&cnt[d], 1.0f);
  }
}

__global__ __launch_bounds__(256)
void divide_mean(float* __restrict__ sums, const float* __restrict__ cnt,
                 long long n, int F) {
  long long i = (long long)blockIdx.x * blockDim.x + threadIdx.x;
  long long s = (long long)gridDim.x * blockDim.x;
  for (; i < n; i += s) sums[i] /= fmaxf(cnt[i / F], 1.0f);
}

__global__ __launch_bounds__(256)
void gather_idx(const int* __restrict__ a, const int* __restrict__ b,
                int* __restrict__ out, int n) {
  int i = blockIdx.x * blockDim.x + threadIdx.x;
  if (i < n) out[i] = b[a[i]];
}

// Per-column sum / sumsq with block-level LDS reduction (grid.y = column)
__global__ __launch_bounds__(256)
void bn_stats(const float* __restrict__ X, float* __restrict__ sum,
              float* __restrict__ sumsq, long long M, int C) {
  int c = blockIdx.y;
  float s = 0.f, s2 = 0.f;
  long long stride = (long long)gridDim.x * blockDim.x;
  for (long long r = (long long)blockIdx.x * blockDim.x + threadIdx.x;
       r < M; r += stride) {
    float v = X[r * C + c];
    s += v; s2 += v * v;
  }
  __shared__ float ls[256], ls2[256];
  ls[threadIdx.x] = s; ls2[threadIdx.x] = s2;
  __syncthreads();
  for (int o = 128; o > 0; o >>= 1) {
    if ((int)threadIdx.x < o) {
      ls[threadIdx.x]  += ls[threadIdx.x + o];
      ls2[threadIdx.x] += ls2[threadIdx.x + o];
    }
    __syncthreads();
  }
  if (threadIdx.x == 0) {
    atomicAdd(&sum[c], ls[0]);
    atomicAdd(&sumsq[c], ls2[0]);
  }
}

__global__ __launch_bounds__(256)
void bn_apply(float* __restrict__ X, const float* __restrict__ sum,
              const float* __restrict__ sumsq, const float* __restrict__ g,
              const float* __restrict__ b, long long M, int C) {
  long long total = M * C;
  long long stride = (long long)gridDim.x * blockDim.x;
  float invM = 1.0f / (float)M;
  for (long long t = (long long)blockIdx.x * blockDim.x + threadIdx.x;
       t < total; t += stride) {
    int c = (int)(t % C);
    float mean = sum[c] * invM;
    float var  = sumsq[c] * invM - mean * mean;
    X[t] = (X[t] - mean) * rsqrtf(var + 1e-5f) * g[c] + b[c];
  }
}

// ---------------- host side ----------------

static inline int gblk(long long n) {
  long long b = (n + 255) / 256;
  return (int)(b > 65535 ? 65535 : b);
}

static unsigned short* g_wbf = nullptr;   // bf16 weight scratch (set per launch)

static void gemm(hipStream_t st,
                 const float* S0, const int* I0, int W0,
                 const float* S1, const int* I1, int W1,
                 const float* S2, const int* I2, int W2,
                 const float* Wm, const float* bias, float* Y,
                 int M, int N, int K, int relu) {
  long long wn = (long long)N * K;
  cvt_f32_bf16<<<gblk(wn), 256, 0, st>>>(Wm, g_wbf, wn);
  dim3 grid((M + TILE_M - 1) / TILE_M, (N + TILE_N - 1) / TILE_N);
  wmma_gemm_gather<<<grid, 256, 0, st>>>(S0, I0, W0, S1, I1, W1, S2, I2, W2,
                                         g_wbf, bias, Y, M, N, K, relu);
}

static void zero(hipStream_t st, float* p, long long n) {
  fill_f32<<<gblk(n), 256, 0, st>>>(p, 0.f, n);
}

static void scatter_mean(hipStream_t st, const float* V, const int* idx,
                         float* out, float* cnt, long long M, int segs, int F) {
  zero(st, out, (long long)segs * F);
  zero(st, cnt, segs);
  scatter_add<<<gblk(M * F), 256, 0, st>>>(V, idx, out, cnt, M, F);
  divide_mean<<<gblk((long long)segs * F), 256, 0, st>>>(out, cnt,
                                                         (long long)segs * F, F);
}

static void batchnorm(hipStream_t st, float* X, const float* g, const float* b,
                      float* ssum, float* ssq, long long M, int C) {
  zero(st, ssum, C);
  zero(st, ssq, C);
  dim3 grid(64, C);
  bn_stats<<<grid, 256, 0, st>>>(X, ssum, ssq, M, C);
  bn_apply<<<gblk(M * C), 256, 0, st>>>(X, ssum, ssq, g, b, M, C);
}

extern "C" void kernel_launch(void* const* d_in, const int* in_sizes, int n_in,
                              void* d_out, int out_size, void* d_ws, size_t ws_size,
                              hipStream_t stream) {
  const float* x     = (const float*)d_in[0];     // [N,64]
  const float* ea    = (const float*)d_in[1];     // [E,16]
  const int*   ei    = (const int*)d_in[2];       // [2,E]
  const int*   batch = (const int*)d_in[3];       // [N]
  const int E  = in_sizes[2] / 2;
  const int Nn = in_sizes[3];
  const int G  = out_size;                        // 256 graphs

  const float* const* P = (const float* const*)d_in;
  // layer param blocks: l1 @5, l2 @21
  const int L1 = 5, L2 = 21;
  // offsets: eW1 0 eb1 1 eW2 2 eb2 3 nW1a 4 nb1a 5 nW1b 6 nb1b 7
  //          nW2a 8 nb2a 9 nW2b 10 nb2b 11 gW1 12 gb1 13 gW2 14 gb2 15
  const float* node_g = P[37]; const float* node_b = P[38];
  const float* edge_g = P[39]; const float* edge_b = P[40];
  const float* u_g    = P[41]; const float* u_b    = P[42];
  const float* fc1W = P[43]; const float* fc1b = P[44];
  const float* fc2W = P[45]; const float* fc2b = P[46];

  const int* row = ei;
  const int* col = ei + E;

  // workspace carve (floats)
  float* w = (float*)d_ws;
  size_t off = 0;
  auto take = [&](long long n) { float* p = w + off; off += (size_t)n; return p; };
  float* bufA  = take((long long)E * 128);
  float* bufB  = take((long long)E * 128);
  float* e_buf = take((long long)E * 128);
  float* nbuf  = take((long long)Nn * 128);
  float* x1    = take((long long)Nn * 256);
  float* agg   = take((long long)Nn * 128);
  float* cnt   = take(Nn);
  int*   bRow  = (int*)take(E);
  float* eg    = take((long long)G * 128);
  float* gcnt  = take(G);
  float* u0    = take((long long)G * 64);
  float* u1    = take((long long)G * 64);
  float* sbuf  = take((long long)G * 64);
  float* u2    = take((long long)G * 64);
  float* hbuf  = take((long long)G * 16);
  float* ssum  = take(256);
  float* ssq   = take(256);
  g_wbf        = (unsigned short*)take(41000);    // bf16 weights (max 128x640)
  (void)ws_size; (void)n_in;

  // ---------------- Layer 1 ----------------
  // EdgeModel: e1 = MLP(cat[x[row], x[col], ea])
  gemm(stream, x, row, 64, x, col, 64, ea, nullptr, 16,
       P[L1+0], P[L1+1], bufA, E, 64, 144, 1);
  gemm(stream, bufA, nullptr, 64, nullptr, nullptr, 0, nullptr, nullptr, 0,
       P[L1+2], P[L1+3], e_buf, E, 128, 64, 0);
  // NodeModel: m = MLP(cat[x[col], e1]); agg = scatter_mean(m, row)
  gemm(stream, x, col, 64, e_buf, nullptr, 128, nullptr, nullptr, 0,
       P[L1+4], P[L1+5], bufB, E, 128, 192, 1);
  gemm(stream, bufB, nullptr, 128, nullptr, nullptr, 0, nullptr, nullptr, 0,
       P[L1+6], P[L1+7], bufA, E, 128, 128, 0);
  scatter_mean(stream, bufA, row, agg, cnt, E, Nn, 128);
  // xn = MLP(cat[x, agg])
  gemm(stream, x, nullptr, 64, agg, nullptr, 128, nullptr, nullptr, 0,
       P[L1+8], P[L1+9], nbuf, Nn, 128, 192, 1);
  gemm(stream, nbuf, nullptr, 128, nullptr, nullptr, 0, nullptr, nullptr, 0,
       P[L1+10], P[L1+11], x1, Nn, 256, 128, 0);
  // GlobalModel: eg = scatter_mean(e1, batch[row]); u1 = MLP(cat[u0, eg])
  gather_idx<<<(E + 255) / 256, 256, 0, stream>>>(row, batch, bRow, E);
  scatter_mean(stream, e_buf, bRow, eg, gcnt, E, G, 128);
  zero(stream, u0, (long long)G * 64);
  gemm(stream, u0, nullptr, 64, eg, nullptr, 128, nullptr, nullptr, 0,
       P[L1+12], P[L1+13], sbuf, G, 64, 192, 1);
  gemm(stream, sbuf, nullptr, 64, nullptr, nullptr, 0, nullptr, nullptr, 0,
       P[L1+14], P[L1+15], u1, G, 64, 64, 0);
  // BatchNorms (training-mode batch stats, biased variance)
  batchnorm(stream, x1, node_g, node_b, ssum, ssq, Nn, 256);
  batchnorm(stream, e_buf, edge_g, edge_b, ssum, ssq, E, 128);
  batchnorm(stream, u1, u_g, u_b, ssum, ssq, G, 64);

  // -------- Layer 2 (NodeModel is dead code for the output -> skipped) --------
  gemm(stream, x1, row, 256, x1, col, 256, e_buf, nullptr, 128,
       P[L2+0], P[L2+1], bufA, E, 128, 640, 1);
  gemm(stream, bufA, nullptr, 128, nullptr, nullptr, 0, nullptr, nullptr, 0,
       P[L2+2], P[L2+3], bufB, E, 128, 128, 0);
  scatter_mean(stream, bufB, bRow, eg, gcnt, E, G, 128);
  gemm(stream, u1, nullptr, 64, eg, nullptr, 128, nullptr, nullptr, 0,
       P[L2+12], P[L2+13], sbuf, G, 64, 192, 1);
  gemm(stream, sbuf, nullptr, 64, nullptr, nullptr, 0, nullptr, nullptr, 0,
       P[L2+14], P[L2+15], u2, G, 64, 64, 0);

  // ---------------- head: fc1 -> relu -> fc2 ----------------
  gemm(stream, u2, nullptr, 64, nullptr, nullptr, 0, nullptr, nullptr, 0,
       fc1W, fc1b, hbuf, G, 16, 64, 1);
  gemm(stream, hbuf, nullptr, 16, nullptr, nullptr, 0, nullptr, nullptr, 0,
       fc2W, fc2b, (float*)d_out, G, 1, 16, 0);
}
